// LinearAttention_67774583931296
// MI455X (gfx1250) — matrix operand order repaired and checked by reading, not compiled, not run on hardware
//
#include <hip/hip_runtime.h>

typedef __attribute__((ext_vector_type(16))) _Float16 v16h;
typedef __attribute__((ext_vector_type(8)))  _Float16 v8h;
typedef __attribute__((ext_vector_type(8)))  float    v8f;

union AF { v16h v; v8h h8[2]; };
union CF { v8f  v; float f[8]; };
union HP { unsigned int u; _Float16 h[2]; };

#define N_HEADS  4
#define D_HEAD   32
#define C_DIM    128
#define N_BATCH  8
#define NPIX     16384   /* 128*128 */

__device__ inline v8f v8f_zero() {
  CF u;
#pragma unroll
  for (int i = 0; i < 8; ++i) u.f[i] = 0.0f;
  return u.v;
}

// A fragment: 16x32 f16 tile of a row-major matrix (stride in elements).
// Lane l<16 : M=l,   K in {k0+0..7,  k0+16..23}
// Lane l>=16: M=l-16, K in {k0+8..15, k0+24..31}
__device__ inline v16h load_frag_a(const _Float16* p, int row0, int k0, int stride, int lane) {
  int sel = lane >> 4, m = lane & 15;
  const _Float16* base = p + (row0 + m) * stride + k0 + sel * 8;
  AF a;
  a.h8[0] = *(const v8h*)(base);
  a.h8[1] = *(const v8h*)(base + 16);
  return a.v;
}

// B fragment: 32x16 f16 tile of B[k][n], where storage is n-major: T[n][k]
__device__ inline v16h load_frag_b(const _Float16* p, int n0, int k0, int stride, int lane) {
  int sel = lane >> 4, col = lane & 15;
  const _Float16* base = p + (n0 + col) * stride + k0 + sel * 8;
  AF a;
  a.h8[0] = *(const v8h*)(base);
  a.h8[1] = *(const v8h*)(base + 16);
  return a.v;
}

// stage a 64(n) x 128(c) fp32 tile from global (c-major) to LDS n-major f16.
// Each thread: 2 float4 loads (two adjacent channels), 4 packed u32 DS stores.
__device__ inline void stage_x_tile(const float* __restrict__ x, _Float16* xT,
                                    int b, int n0g, int t) {
#pragma unroll
  for (int i = 0; i < 4; ++i) {
    int p = i * 256 + t;            // 0..1023
    int c = (p >> 4) * 2;           // even channel 0..126
    int j0 = (p & 15) * 4;          // col group 0..60
    const float4 va = *(const float4*)(x + ((size_t)(b * C_DIM + c) << 14) + n0g + j0);
    const float4 vb = *(const float4*)(x + ((size_t)(b * C_DIM + c + 1) << 14) + n0g + j0);
    HP h0, h1, h2, h3;
    h0.h[0] = (_Float16)va.x; h0.h[1] = (_Float16)vb.x;
    h1.h[0] = (_Float16)va.y; h1.h[1] = (_Float16)vb.y;
    h2.h[0] = (_Float16)va.z; h2.h[1] = (_Float16)vb.z;
    h3.h[0] = (_Float16)va.w; h3.h[1] = (_Float16)vb.w;
    *(unsigned int*)(xT + (j0 + 0) * C_DIM + c) = h0.u;
    *(unsigned int*)(xT + (j0 + 1) * C_DIM + c) = h1.u;
    *(unsigned int*)(xT + (j0 + 2) * C_DIM + c) = h2.u;
    *(unsigned int*)(xT + (j0 + 3) * C_DIM + c) = h3.u;
  }
}

// ---------------- K0: convert Wq|Wk rows (0..255) of Wqkv to f16 ----------------
__global__ __launch_bounds__(256) void convert_wqk_kernel(const float* __restrict__ Wqkv,
                                                          _Float16* __restrict__ wqk) {
  int i = blockIdx.x * 256 + threadIdx.x;
  if (i < 256 * C_DIM) wqk[i] = (_Float16)Wqkv[i];
}

// ---------------- K1: q,k = Wqk @ x (256x64 tile per block), fused q-softmax ----
__global__ __launch_bounds__(256) void qk_gemm_kernel(const float* __restrict__ x,
                                                      const _Float16* __restrict__ wqk,
                                                      _Float16* __restrict__ qs,
                                                      _Float16* __restrict__ kbuf) {
  extern __shared__ char smem[];
  _Float16* xT = (_Float16*)smem;              // [64][128] n-major, f16
  int b = blockIdx.y;
  int n0g = blockIdx.x * 64;
  int t = threadIdx.x;

  stage_x_tile(x, xT, b, n0g, t);
  __syncthreads();

  int w = t >> 5, lane = t & 31;
  v8f acc[2][4];
#pragma unroll
  for (int rt = 0; rt < 2; ++rt)
#pragma unroll
    for (int nt = 0; nt < 4; ++nt) acc[rt][nt] = v8f_zero();

#pragma unroll
  for (int kk = 0; kk < C_DIM; kk += 32) {
    v16h afr[2], bfr[4];
#pragma unroll
    for (int rt = 0; rt < 2; ++rt)
      afr[rt] = load_frag_a(wqk, w * 32 + rt * 16, kk, C_DIM, lane);
#pragma unroll
    for (int nt = 0; nt < 4; ++nt)
      bfr[nt] = load_frag_b(xT, nt * 16, kk, C_DIM, lane);
#pragma unroll
    for (int rt = 0; rt < 2; ++rt)
#pragma unroll
      for (int nt = 0; nt < 4; ++nt)
        acc[rt][nt] = __builtin_amdgcn_wmma_f32_16x16x32_f16(
            false, afr[rt], false, bfr[nt], (short)0, acc[rt][nt], false, false);
  }

  int sel = lane >> 4, cl = lane & 15;
  if (w < 4) {
    // wave w == head w; softmax over d=32 of one column: my 16 values + lane^16.
    const float scale = 0.17677669529663687f;  // 32^-0.5
#pragma unroll
    for (int nt = 0; nt < 4; ++nt) {
      CF c0, c1; c0.v = acc[0][nt]; c1.v = acc[1][nt];
      float mx = -1e30f;
#pragma unroll
      for (int r = 0; r < 8; ++r) { mx = fmaxf(mx, c0.f[r]); mx = fmaxf(mx, c1.f[r]); }
      mx = fmaxf(mx, __shfl_xor(mx, 16, 32));
      float e0[8], e1[8];
      float s = 0.f;
#pragma unroll
      for (int r = 0; r < 8; ++r) {
        e0[r] = __expf(c0.f[r] - mx);
        e1[r] = __expf(c1.f[r] - mx);
        s += e0[r] + e1[r];
      }
      s += __shfl_xor(s, 16, 32);
      float inv = scale / s;
      int n = n0g + nt * 16 + cl;
      size_t base = ((size_t)(b * N_HEADS + w) * D_HEAD) << 14;
#pragma unroll
      for (int r = 0; r < 8; ++r) {
        int d0 = r + sel * 8;
        int d1 = 16 + r + sel * 8;
        qs[base + ((size_t)d0 << 14) + n] = (_Float16)(e0[r] * inv);
        qs[base + ((size_t)d1 << 14) + n] = (_Float16)(e1[r] * inv);
      }
    }
  } else {
    int ck0 = (w - 4) * 32;
#pragma unroll
    for (int nt = 0; nt < 4; ++nt) {
      CF c0, c1; c0.v = acc[0][nt]; c1.v = acc[1][nt];
      int n = n0g + nt * 16 + cl;
#pragma unroll
      for (int r = 0; r < 8; ++r) {
        int row0 = ck0 + r + sel * 8;
        int row1 = ck0 + 16 + r + sel * 8;
        kbuf[((size_t)(b * C_DIM + row0) << 14) + n] = (_Float16)c0.f[r];
        kbuf[((size_t)(b * C_DIM + row1) << 14) + n] = (_Float16)c1.f[r];
      }
    }
  }
}

// ---------------- K2: softmax of k over n; async LDS staging (ASYNCcnt path) ----
__global__ __launch_bounds__(256) void ksoftmax_kernel(_Float16* __restrict__ kbuf) {
  extern __shared__ char smem[];
  _Float16* krow = (_Float16*)smem;            // 16384 halfs = 32KB
  float* red = (float*)(smem + 32768);         // 256 floats
  int row = blockIdx.x;                        // 0..1023
  size_t base = (size_t)row << 14;
  int t = threadIdx.x;

  // async copy: global row -> LDS (tracked by ASYNCcnt, no VGPR landing)
#pragma unroll
  for (int i = 0; i < 8; ++i) {
    int e = (i * 256 + t) * 8;                 // 8 halfs (16B) per op
    unsigned int lo = (unsigned int)(size_t)(krow + e);  // low 32b = LDS offset
    const _Float16* g = kbuf + base + e;
    asm volatile("global_load_async_to_lds_b128 %0, %1, off"
                 :: "v"(lo), "v"(g) : "memory");
  }
  asm volatile("s_wait_asynccnt 0" ::: "memory");
  __syncthreads();

  // cache this thread's 64 halfs in registers
  v8h vals[8];
#pragma unroll
  for (int i = 0; i < 8; ++i) vals[i] = *(const v8h*)(krow + (i * 256 + t) * 8);

  float mx = -1e30f;
#pragma unroll
  for (int i = 0; i < 8; ++i)
#pragma unroll
    for (int q = 0; q < 8; ++q) mx = fmaxf(mx, (float)vals[i][q]);
  red[t] = mx; __syncthreads();
  for (int s = 128; s > 0; s >>= 1) { if (t < s) red[t] = fmaxf(red[t], red[t + s]); __syncthreads(); }
  mx = red[0]; __syncthreads();

  float sum = 0.f;
#pragma unroll
  for (int i = 0; i < 8; ++i)
#pragma unroll
    for (int q = 0; q < 8; ++q) sum += __expf((float)vals[i][q] - mx);
  red[t] = sum; __syncthreads();
  for (int s = 128; s > 0; s >>= 1) { if (t < s) red[t] += red[t + s]; __syncthreads(); }
  float inv = 1.f / red[0];

  // normalize into LDS, then async store LDS -> global
#pragma unroll
  for (int i = 0; i < 8; ++i) {
    v8h o;
#pragma unroll
    for (int q = 0; q < 8; ++q) o[q] = (_Float16)(__expf((float)vals[i][q] - mx) * inv);
    *(v8h*)(krow + (i * 256 + t) * 8) = o;
  }
  asm volatile("s_wait_dscnt 0" ::: "memory");   // LDS writes visible before async read
#pragma unroll
  for (int i = 0; i < 8; ++i) {
    int e = (i * 256 + t) * 8;
    unsigned int lo = (unsigned int)(size_t)(krow + e);
    _Float16* g = kbuf + base + e;
    asm volatile("global_store_async_from_lds_b128 %0, %1, off"
                 :: "v"(g), "v"(lo) : "memory");
  }
  asm volatile("s_wait_asynccnt 0" ::: "memory");
}

// ---------------- K3: partial A = qs @ ks^T per (b,h), split-K over 8 blocks ----
__global__ __launch_bounds__(256) void attn_a_kernel(const _Float16* __restrict__ qs,
                                                     const _Float16* __restrict__ ks,
                                                     float* __restrict__ Apart) {
  extern __shared__ char smem[];
  float* red = (float*)smem;           // 8 waves * 1024 floats = 32KB
  int bh = blockIdx.x;                 // 0..31
  int chunk = blockIdx.y;              // 0..7
  const _Float16* qsb = qs + ((size_t)(32 * bh) << 14);
  const _Float16* ksb = ks + ((size_t)(32 * bh) << 14);
  int t = threadIdx.x, w = t >> 5, lane = t & 31;

  v8f acc[2][2];
#pragma unroll
  for (int i = 0; i < 2; ++i)
#pragma unroll
    for (int j = 0; j < 2; ++j) acc[i][j] = v8f_zero();

  int nbase = chunk * 2048 + w * 256;
#pragma unroll
  for (int kk = 0; kk < 256; kk += 32) {
    int k0 = nbase + kk;
    v16h a0 = load_frag_a(qsb, 0, k0, NPIX, lane);
    v16h a1 = load_frag_a(qsb, 16, k0, NPIX, lane);
    v16h b0 = load_frag_b(ksb, 0, k0, NPIX, lane);
    v16h b1 = load_frag_b(ksb, 16, k0, NPIX, lane);
    acc[0][0] = __builtin_amdgcn_wmma_f32_16x16x32_f16(false, a0, false, b0, (short)0, acc[0][0], false, false);
    acc[0][1] = __builtin_amdgcn_wmma_f32_16x16x32_f16(false, a0, false, b1, (short)0, acc[0][1], false, false);
    acc[1][0] = __builtin_amdgcn_wmma_f32_16x16x32_f16(false, a1, false, b0, (short)0, acc[1][0], false, false);
    acc[1][1] = __builtin_amdgcn_wmma_f32_16x16x32_f16(false, a1, false, b1, (short)0, acc[1][1], false, false);
  }
#pragma unroll
  for (int i = 0; i < 4; ++i) {
    CF u; u.v = acc[i >> 1][i & 1];
#pragma unroll
    for (int r = 0; r < 8; ++r) red[w * 1024 + i * 256 + lane * 8 + r] = u.f[r];
  }
  __syncthreads();
#pragma unroll
  for (int j = 0; j < 4; ++j) {
    int idx = j * 256 + t;             // 0..1023
    float s = 0.f;
#pragma unroll
    for (int ww = 0; ww < 8; ++ww) s += red[ww * 1024 + idx];
    int tile = idx >> 8, li = (idx >> 3) & 31, r = idx & 7;
    int m = (tile >> 1) * 16 + r + 8 * (li >> 4);
    int e = (tile & 1) * 16 + (li & 15);
    Apart[((size_t)chunk * 32 + bh) * 1024 + m * 32 + e] = s;
  }
}

// ---------------- K4: W3_b = Wout @ blockdiag(A_b) @ Wv (per batch) -------------
__global__ __launch_bounds__(256) void w3_kernel(const float* __restrict__ Wout,
                                                 const float* __restrict__ Wqkv,
                                                 const float* __restrict__ Apart,
                                                 _Float16* __restrict__ W3) {
  extern __shared__ char smem[];
  float* Al = (float*)smem;       // 4*32*32 = 4096 floats
  float* W2 = Al + 4096;          // 128*128 = 16384 floats
  int b = blockIdx.x, t = threadIdx.x;
  for (int i = t; i < 4096; i += 256) {
    int h = i >> 10, inner = i & 1023;
    float s = 0.f;
#pragma unroll
    for (int ch = 0; ch < 8; ++ch)
      s += Apart[((size_t)ch * 32 + b * 4 + h) * 1024 + inner];
    Al[i] = s;
  }
  __syncthreads();
  for (int i = t; i < 16384; i += 256) {
    int o = i >> 7, j = i & 127, h = j >> 5, e = j & 31;
    float s = 0.f;
#pragma unroll
    for (int d = 0; d < 32; ++d)
      s += Wout[o * C_DIM + h * 32 + d] * Al[h * 1024 + d * 32 + e];
    W2[o * C_DIM + j] = s;
  }
  __syncthreads();
  for (int i = t; i < 16384; i += 256) {
    int o = i >> 7, c = i & 127;
    float s = 0.f;
    for (int j = 0; j < 128; ++j)
      s += W2[o * C_DIM + j] * Wqkv[(size_t)(256 + j) * C_DIM + c];
    W3[(size_t)b * 16384 + i] = (_Float16)s;
  }
}

// ---------------- K5: y = W3_b @ x + bout; partial layernorm sums ---------------
__global__ __launch_bounds__(256) void y_gemm_kernel(const float* __restrict__ x,
                                                     const _Float16* __restrict__ W3,
                                                     const float* __restrict__ bout,
                                                     float* __restrict__ y,
                                                     float* __restrict__ partials) {
  extern __shared__ char smem[];
  _Float16* xT = (_Float16*)smem;                   // 64*128 f16 = 16KB
  float* red = (float*)(smem + 64 * C_DIM * 2);     // 512 floats
  int b = blockIdx.y, n0g = blockIdx.x * 64, t = threadIdx.x;
  stage_x_tile(x, xT, b, n0g, t);
  __syncthreads();
  int w = t >> 5, lane = t & 31;
  const _Float16* w3b = W3 + (size_t)b * 16384;
  int rbase = (w & 3) * 32;
  int cbase = (w >> 2) * 32;
  v8f acc[2][2];
#pragma unroll
  for (int i = 0; i < 2; ++i)
#pragma unroll
    for (int j = 0; j < 2; ++j) acc[i][j] = v8f_zero();
#pragma unroll
  for (int kk = 0; kk < C_DIM; kk += 32) {
    v16h a0 = load_frag_a(w3b, rbase, kk, C_DIM, lane);
    v16h a1 = load_frag_a(w3b, rbase + 16, kk, C_DIM, lane);
    v16h b0 = load_frag_b(xT, cbase, kk, C_DIM, lane);
    v16h b1 = load_frag_b(xT, cbase + 16, kk, C_DIM, lane);
    acc[0][0] = __builtin_amdgcn_wmma_f32_16x16x32_f16(false, a0, false, b0, (short)0, acc[0][0], false, false);
    acc[0][1] = __builtin_amdgcn_wmma_f32_16x16x32_f16(false, a0, false, b1, (short)0, acc[0][1], false, false);
    acc[1][0] = __builtin_amdgcn_wmma_f32_16x16x32_f16(false, a1, false, b0, (short)0, acc[1][0], false, false);
    acc[1][1] = __builtin_amdgcn_wmma_f32_16x16x32_f16(false, a1, false, b1, (short)0, acc[1][1], false, false);
  }
  int sel = lane >> 4, cl = lane & 15;
  float lsum = 0.f, lsq = 0.f;
#pragma unroll
  for (int rt = 0; rt < 2; ++rt)
#pragma unroll
    for (int nt = 0; nt < 2; ++nt) {
      CF u; u.v = acc[rt][nt];
#pragma unroll
      for (int r = 0; r < 8; ++r) {
        int o = rbase + rt * 16 + r + 8 * sel;
        int n = n0g + cbase + nt * 16 + cl;
        float yy = u.f[r] + bout[o];
        y[((size_t)(b * C_DIM + o) << 14) + n] = yy;
        lsum += yy; lsq += yy * yy;
      }
    }
  red[t] = lsum; red[256 + t] = lsq;
  __syncthreads();
  for (int s = 128; s > 0; s >>= 1) {
    if (t < s) { red[t] += red[t + s]; red[256 + t] += red[256 + t + s]; }
    __syncthreads();
  }
  if (t == 0) {
    int pb = b * 256 + blockIdx.x;
    partials[pb * 2] = red[0];
    partials[pb * 2 + 1] = red[256];
  }
}

// ---------------- K6: finalize per-batch mean / inv-std -------------------------
__global__ __launch_bounds__(256) void stats_kernel(const float* __restrict__ partials,
                                                    float* __restrict__ stats) {
  __shared__ float red[512];
  int b = blockIdx.x, t = threadIdx.x;
  red[t] = partials[(b * 256 + t) * 2];
  red[256 + t] = partials[(b * 256 + t) * 2 + 1];
  __syncthreads();
  for (int s = 128; s > 0; s >>= 1) {
    if (t < s) { red[t] += red[t + s]; red[256 + t] += red[256 + t + s]; }
    __syncthreads();
  }
  if (t == 0) {
    const float Ninv = 1.f / (128.f * 16384.f);
    float mean = red[0] * Ninv;
    float var = red[256] * Ninv - mean * mean;
    stats[b * 2] = mean;
    stats[b * 2 + 1] = rsqrtf(var + 1e-5f);
  }
}

// ---------------- K7: normalize + gamma/beta (in place on d_out) ----------------
__global__ __launch_bounds__(256) void norm_kernel(float* __restrict__ y,
                                                   const float* __restrict__ stats,
                                                   const float* __restrict__ gamma,
                                                   const float* __restrict__ beta) {
  size_t idx = (size_t)blockIdx.x * 256 + threadIdx.x;
  int b = (int)(idx >> 21);          // 128*16384 = 2^21 elems per batch
  int c = (int)((idx >> 14) & 127);
  float mean = stats[b * 2], inv = stats[b * 2 + 1];
  y[idx] = (y[idx] - mean) * inv * gamma[c] + beta[c];
}

extern "C" void kernel_launch(void* const* d_in, const int* in_sizes, int n_in,
                              void* d_out, int out_size, void* d_ws, size_t ws_size,
                              hipStream_t stream) {
  (void)in_sizes; (void)n_in; (void)out_size; (void)ws_size;
  const float* x     = (const float*)d_in[0];
  const float* Wqkv  = (const float*)d_in[1];
  const float* Wout  = (const float*)d_in[2];
  const float* bout  = (const float*)d_in[3];
  const float* gamma = (const float*)d_in[4];
  const float* beta  = (const float*)d_in[5];
  float* y = (float*)d_out;

  char* ws = (char*)d_ws;
  size_t off = 0;
  _Float16* qs   = (_Float16*)(ws + off); off += (size_t)N_BATCH * C_DIM * NPIX * 2; // 33.5MB
  _Float16* kbuf = (_Float16*)(ws + off); off += (size_t)N_BATCH * C_DIM * NPIX * 2; // 33.5MB
  _Float16* wqk  = (_Float16*)(ws + off); off += (size_t)256 * C_DIM * 2;
  _Float16* w3   = (_Float16*)(ws + off); off += (size_t)N_BATCH * C_DIM * C_DIM * 2;
  float* Apart    = (float*)(ws + off);   off += (size_t)8 * 32 * 1024 * 4;          // 1MB
  float* partials = (float*)(ws + off);   off += (size_t)2048 * 2 * 4;
  float* stats    = (float*)(ws + off);   off += 64;

  convert_wqk_kernel<<<128, 256, 0, stream>>>(Wqkv, wqk);
  qk_gemm_kernel<<<dim3(256, 8), 256, 64 * C_DIM * 2, stream>>>(x, wqk, qs, kbuf);
  ksoftmax_kernel<<<1024, 256, 32768 + 1024, stream>>>(kbuf);
  attn_a_kernel<<<dim3(32, 8), 256, 8 * 1024 * 4, stream>>>(qs, kbuf, Apart);
  w3_kernel<<<8, 256, (4096 + 16384) * 4, stream>>>(Wout, Wqkv, Apart, w3);
  y_gemm_kernel<<<dim3(256, 8), 256, 64 * C_DIM * 2 + 512 * 4, stream>>>(x, w3, bout, y, partials);
  stats_kernel<<<8, 256, 0, stream>>>(partials, stats);
  norm_kernel<<<65536, 256, 0, stream>>>(y, stats, gamma, beta);
}
